// MLP_40415642256059
// MI455X (gfx1250) — matrix operand order repaired and checked by reading, not compiled, not run on hardware
//
#include <hip/hip_runtime.h>

typedef __attribute__((ext_vector_type(16))) _Float16 v16h;
typedef __attribute__((ext_vector_type(8)))  _Float16 v8h;
typedef __attribute__((ext_vector_type(8)))  float    v8f;
typedef __attribute__((ext_vector_type(4)))  int      v4i;

#define NB 4096   // batch
#define ND 512    // D_IN
#define NH 2048   // H

#if __has_builtin(__builtin_amdgcn_global_load_async_to_lds_b128)
#define HAS_ASYNC_LDS 1
#else
#define HAS_ASYNC_LDS 0
#endif

typedef __attribute__((address_space(1))) v4i gv4i;  // global int4
typedef __attribute__((address_space(3))) v4i lv4i;  // LDS int4

__device__ __forceinline__ void async_copy_b128(const _Float16* g, _Float16* l) {
#if HAS_ASYNC_LDS
  __builtin_amdgcn_global_load_async_to_lds_b128((gv4i*)g, (lv4i*)l, 0, 0);
#else
  *(v8h*)l = *(const v8h*)g;  // fallback when builtin unavailable
#endif
}

__device__ __forceinline__ void wait_async0() {
#if HAS_ASYNC_LDS
#if __has_builtin(__builtin_amdgcn_s_wait_asynccnt)
  __builtin_amdgcn_s_wait_asynccnt(0);
#else
  asm volatile("s_wait_asynccnt 0x0" ::: "memory");
#endif
#endif
}

// ---------------------------------------------------------------- convert f32 -> f16 (x)
__global__ void f32_to_f16_kernel(const float* __restrict__ src,
                                  _Float16* __restrict__ dst, int n) {
  int i = blockIdx.x * blockDim.x + threadIdx.x;
  if (i < n) dst[i] = (_Float16)src[i];
}

// ---------------------------------------------------------------- transpose+convert: W[K][N] -> Wt[N][K] (f16)
__global__ __launch_bounds__(256) void transpose_f32_to_f16_kernel(
    const float* __restrict__ src, _Float16* __restrict__ dst, int K, int N) {
  __shared__ float t[32][33];
  const int nb = blockIdx.x * 32;
  const int kb = blockIdx.y * 32;
  const int tx = threadIdx.x & 31;
  const int ty = threadIdx.x >> 5;  // 0..7
#pragma unroll
  for (int r = ty; r < 32; r += 8)
    t[r][tx] = src[(size_t)(kb + r) * N + nb + tx];
  __syncthreads();
#pragma unroll
  for (int r = ty; r < 32; r += 8)
    dst[(size_t)(nb + r) * K + kb + tx] = (_Float16)t[tx][r];
}

// ---------------------------------------------------------------- tiled WMMA GEMM
// C[M=grid.x*128][2048] = relu(A[M][K] * Wt[n][K]^T + bias)
//   A : [M][K] f16 row-major, Wt : [2048][K] f16 row-major (pre-transposed W)
// FUSE=false: store C as f16 into Hout.
// FUSE=true : fold matvec with W3 -> partialOut[blockIdx.y*4096 + m].
template <int K, bool FUSE>
__global__ __launch_bounds__(256) void gemm_kernel(
    const _Float16* __restrict__ A, const _Float16* __restrict__ Wt,
    const float* __restrict__ bias, _Float16* __restrict__ Hout,
    const float* __restrict__ W3, float* __restrict__ partialOut) {
  constexpr int NN = 2048;
  constexpr int BM = 128, BN = 128, BK = 32;
  constexpr int LDT = BK + 8;  // 40 halves = 80B row (16B multiple)

  __shared__ __align__(16) _Float16 As[2][BM][LDT];  // [m][k]
  __shared__ __align__(16) _Float16 Bs[2][BN][LDT];  // [n][k]
  __shared__ float sW[2][BM];

  const int tid   = threadIdx.x;
  const int lane  = tid & 31;
  const int wid   = tid >> 5;   // 8 wave32 waves
  const int waveM = wid & 3;    // 4 wave-rows * 32 m
  const int waveN = wid >> 2;   // 2 wave-cols * 64 n
  const int mWG   = blockIdx.x * BM;
  const int nWG   = blockIdx.y * BN;

  // staging map: 256 threads, 16 halves (2 x b128) per operand each
  const int sr = tid >> 1;        // 0..127 (row within tile)
  const int sc = (tid & 1) * 16;  // 0/16   (k-chunk)

  const _Float16* Ap = A + (size_t)(mWG + sr) * K + sc;
  const _Float16* Bp = Wt + (size_t)(nWG + sr) * K + sc;

  v8f acc[2][4];
#pragma unroll
  for (int mt = 0; mt < 2; ++mt)
#pragma unroll
    for (int nt = 0; nt < 4; ++nt)
      acc[mt][nt] = (v8f){0.f, 0.f, 0.f, 0.f, 0.f, 0.f, 0.f, 0.f};

  const int arow = lane & 15;
  const int kHi  = (lane >> 4) * 8;   // A: K pattern {kHi..kHi+7, kHi+16..kHi+23}
  const int kB   = (lane >> 4) * 8;   // B: K pattern {kB..kB+7, kB+16..kB+23}

  auto stage = [&](int b, int k0) {
    async_copy_b128(Ap + k0,     &As[b][sr][sc]);
    async_copy_b128(Ap + k0 + 8, &As[b][sr][sc + 8]);
    async_copy_b128(Bp + k0,     &Bs[b][sr][sc]);
    async_copy_b128(Bp + k0 + 8, &Bs[b][sr][sc + 8]);
  };

  // prologue: stage tile 0
  stage(0, 0);
  wait_async0();
  __syncthreads();

  int buf = 0;
  for (int k0 = 0; k0 < K; k0 += BK) {
    if (k0 + BK < K) stage(buf ^ 1, k0 + BK);  // prefetch next tile (async)

    v16h af[2], bf[4];
#pragma unroll
    for (int mt = 0; mt < 2; ++mt) {
      int r = waveM * 32 + mt * 16 + arow;
      v8h lo = *(const v8h*)&As[buf][r][kHi];
      v8h hi = *(const v8h*)&As[buf][r][kHi + 16];
      af[mt] = __builtin_shufflevector(lo, hi, 0, 1, 2, 3, 4, 5, 6, 7, 8, 9, 10,
                                       11, 12, 13, 14, 15);
    }
#pragma unroll
    for (int nt = 0; nt < 4; ++nt) {
      int c = waveN * 64 + nt * 16 + arow;
      v8h lo = *(const v8h*)&Bs[buf][c][kB];
      v8h hi = *(const v8h*)&Bs[buf][c][kB + 16];
      bf[nt] = __builtin_shufflevector(lo, hi, 0, 1, 2, 3, 4, 5, 6, 7, 8, 9, 10,
                                       11, 12, 13, 14, 15);
    }
#pragma unroll
    for (int mt = 0; mt < 2; ++mt)
#pragma unroll
      for (int nt = 0; nt < 4; ++nt)
        acc[mt][nt] = __builtin_amdgcn_wmma_f32_16x16x32_f16(
            false, af[mt], false, bf[nt], (short)0, acc[mt][nt], false, false);

    wait_async0();     // our prefetch into buf^1 landed in LDS
    __syncthreads();   // everyone finished reading buf & writing buf^1
    buf ^= 1;
  }

  // C/D layout: VGPR v: m = v + 8*(lane>=16), n = lane&15 (per 16x16 tile)
  const int cn = lane & 15;
  const int cm = 8 * (lane >> 4);

  if constexpr (!FUSE) {
#pragma unroll
    for (int mt = 0; mt < 2; ++mt)
#pragma unroll
      for (int nt = 0; nt < 4; ++nt) {
        int nG   = nWG + waveN * 64 + nt * 16 + cn;
        float bv = bias[nG];
#pragma unroll
        for (int v = 0; v < 8; ++v) {
          int mG    = mWG + waveM * 32 + mt * 16 + cm + v;
          float val = acc[mt][nt][v] + bv;
          val       = val > 0.f ? val : 0.f;
          Hout[(size_t)mG * NN + nG] = (_Float16)val;
        }
      }
  } else {
    float pv[2][8];
#pragma unroll
    for (int mt = 0; mt < 2; ++mt)
#pragma unroll
      for (int v = 0; v < 8; ++v) pv[mt][v] = 0.f;
#pragma unroll
    for (int mt = 0; mt < 2; ++mt)
#pragma unroll
      for (int nt = 0; nt < 4; ++nt) {
        int nG   = nWG + waveN * 64 + nt * 16 + cn;
        float bv = bias[nG];
        float w3 = W3[nG];
#pragma unroll
        for (int v = 0; v < 8; ++v) {
          float val = acc[mt][nt][v] + bv;
          val       = val > 0.f ? val : 0.f;
          pv[mt][v] += val * w3;
        }
      }
    // deterministic tree-reduction across the 16 lanes sharing each m row
#pragma unroll
    for (int mt = 0; mt < 2; ++mt)
#pragma unroll
      for (int v = 0; v < 8; ++v) {
        float sv = pv[mt][v];
        sv += __shfl_xor(sv, 1);
        sv += __shfl_xor(sv, 2);
        sv += __shfl_xor(sv, 4);
        sv += __shfl_xor(sv, 8);
        if (cn == 0) sW[waveN][waveM * 32 + mt * 16 + cm + v] = sv;
      }
    __syncthreads();
    for (int i = tid; i < BM; i += 256)
      partialOut[(size_t)blockIdx.y * NB + mWG + i] = sW[0][i] + sW[1][i];
  }
}

// ---------------------------------------------------------------- scores = b3 + sum(partials)
__global__ void reduce_scores_kernel(const float* __restrict__ partial,
                                     const float* __restrict__ b3,
                                     float* __restrict__ scores) {
  int m = blockIdx.x * blockDim.x + threadIdx.x;  // 4096 threads
  float acc = b3[0];
#pragma unroll
  for (int q = 0; q < 16; ++q) acc += partial[q * NB + m];
  scores[m] = acc;
}

// ---------------------------------------------------------------- stable descending rank position
__global__ void rank_pos_kernel(const float* __restrict__ scores,
                                float* __restrict__ s_sorted,
                                int* __restrict__ pos) {
  __shared__ float z[NB];
  int i = blockIdx.x * blockDim.x + threadIdx.x;
  for (int j = threadIdx.x; j < NB; j += blockDim.x) z[j] = scores[j];
  __syncthreads();
  float zi = z[i];
  int p = 0;
  for (int j = 0; j < NB; ++j) {
    float zj = z[j];
    p += (zj > zi) || (zj == zi && j < i);  // stable argsort(-z) position
  }
  pos[i] = p;
  s_sorted[p] = zi;  // pos is a permutation -> race-free scatter
}

// ---------------------------------------------------------------- PAV isotonic (non-increasing), 1 block
__global__ void pav_kernel(const float* __restrict__ s_in,
                           float* __restrict__ primal) {
  const int n = NB;
  __shared__ float s[NB];
  __shared__ float psum[NB];
  __shared__ int   pcnt[NB];
  for (int i = threadIdx.x; i < n; i += blockDim.x) s[i] = s_in[i];
  __syncthreads();
  if (threadIdx.x == 0) {
    int top = 0;
    for (int p = 0; p < n; ++p) {
      float cs = s[p] - (float)(n - p);  // y[p] = s[p] - w[p], w[p] = n-p
      int cc = 1;
      while (top > 0 && cs * (float)pcnt[top - 1] > psum[top - 1] * (float)cc) {
        cs += psum[top - 1];
        cc += pcnt[top - 1];
        --top;
      }
      psum[top] = cs;
      pcnt[top] = cc;
      ++top;
    }
    int p = 0;
    for (int q = 0; q < top; ++q) {
      float mean = psum[q] / (float)pcnt[q];
      int c = pcnt[q];
      for (int t = 0; t < c; ++t) {
        s[p] = s[p] - mean;  // primal = s - dual
        ++p;
      }
    }
  }
  __syncthreads();
  for (int i = threadIdx.x; i < n; i += blockDim.x) primal[i] = s[i];
}

// ---------------------------------------------------------------- gather + emit outputs
__global__ void finalize_kernel(const float* __restrict__ primal,
                                const int* __restrict__ pos,
                                const float* __restrict__ scores,
                                float* __restrict__ out) {
  int i = blockIdx.x * blockDim.x + threadIdx.x;
  out[i] = primal[pos[i]];  // rank, undo the sort
  out[NB + i] = scores[i];  // scores
}

// ---------------------------------------------------------------- launcher
extern "C" void kernel_launch(void* const* d_in, const int* in_sizes, int n_in,
                              void* d_out, int out_size, void* d_ws,
                              size_t ws_size, hipStream_t stream) {
  const float* x  = (const float*)d_in[0];  // [4096,512]
  const float* W1 = (const float*)d_in[1];  // [512,2048]
  const float* b1 = (const float*)d_in[2];  // [2048]
  const float* W2 = (const float*)d_in[3];  // [2048,2048]
  const float* b2 = (const float*)d_in[4];  // [2048]
  const float* W3 = (const float*)d_in[5];  // [2048]
  const float* b3 = (const float*)d_in[6];  // [1]
  float* out = (float*)d_out;               // [4096 rank] ++ [4096 scores]

  uint8_t* w = (uint8_t*)d_ws;
  _Float16* Xh   = (_Float16*)(w);             // 4096*512*2  = 4 MiB
  _Float16* W1t  = (_Float16*)(w + 4194304);   // [2048][512]  f16, 2 MiB
  _Float16* W2t  = (_Float16*)(w + 6291456);   // [2048][2048] f16, 8 MiB
  _Float16* H1h  = (_Float16*)(w + 14680064);  // [4096][2048] f16, 16 MiB
  float* partial = (float*)(w + 31457280);     // 16*4096*4
  float* scores  = (float*)(w + 31719424);     // 4096*4
  float* ssorted = (float*)(w + 31735808);     // 4096*4
  float* primal  = (float*)(w + 31752192);     // 4096*4
  int*   pos     = (int*)(w + 31768576);       // 4096*4

  f32_to_f16_kernel<<<(NB * ND) / 256, 256, 0, stream>>>(x, Xh, NB * ND);
  transpose_f32_to_f16_kernel<<<dim3(NH / 32, ND / 32), 256, 0, stream>>>(
      W1, W1t, ND, NH);
  transpose_f32_to_f16_kernel<<<dim3(NH / 32, NH / 32), 256, 0, stream>>>(
      W2, W2t, NH, NH);

  dim3 grid(NB / 128, NH / 128);  // 32 x 16
  gemm_kernel<ND, false><<<grid, 256, 0, stream>>>(Xh, W1t, b1, H1h, nullptr,
                                                   nullptr);
  gemm_kernel<NH, true><<<grid, 256, 0, stream>>>(H1h, W2t, b2, nullptr, W3,
                                                  partial);

  reduce_scores_kernel<<<NB / 256, 256, 0, stream>>>(partial, b3, scores);
  rank_pos_kernel<<<NB / 256, 256, 0, stream>>>(scores, ssorted, pos);
  pav_kernel<<<1, 256, 0, stream>>>(ssorted, primal);
  finalize_kernel<<<NB / 256, 256, 0, stream>>>(primal, pos, scores, out);
}
